// MegatronBART_44530220925727
// MI455X (gfx1250) — matrix operand order repaired and checked by reading, not compiled, not run on hardware
//
#include <hip/hip_runtime.h>
#include <hip/hip_bf16.h>
#include <cstdint>
#include <cstddef>

// ---------------------------------------------------------------------------
// MegatronBART forward for gfx1250 (MI455X): all GEMMs through
// v_wmma_f32_16x16x32_f16 (f16 in / f32 accumulate), wave32, double-buffered
// LDS with register-staged prefetch: global loads of tile k+1 are issued
// before the WMMAs of tile k, and only converted/stored to LDS afterwards,
// so VMEM latency hides under matrix work.
// ---------------------------------------------------------------------------

constexpr int kL  = 6;
constexpr int kD  = 1024;
constexpr int kH  = 16;
constexpr int kHD = 64;          // head dim
constexpr int kFF = 4096;
constexpr int kS  = 512;         // seq len (src == tgt here)
constexpr int kB  = 8;           // batch
constexpr int kBH = kB * kH;     // 128 head-batches
constexpr int kM  = kS * kB;     // 4096 token rows
constexpr float kEps    = 1e-5f;
constexpr float kQScale = 0.125f; // HD^-0.5

constexpr int FLAG_BIAS = 1;
constexpr int FLAG_GELU = 2;
constexpr int FLAG_RES  = 4;

typedef _Float16 v16h __attribute__((ext_vector_type(16)));
typedef float    v8f  __attribute__((ext_vector_type(8)));
union Frag { v16h v; uint4 u[2]; };

// ---------------------------------------------------------------------------
// Wave-level reductions (wave32)
// ---------------------------------------------------------------------------
__device__ __forceinline__ float waveSum(float v) {
#pragma unroll
  for (int o = 16; o > 0; o >>= 1) v += __shfl_xor(v, o, 32);
  return v;
}
__device__ __forceinline__ float waveMax(float v) {
#pragma unroll
  for (int o = 16; o > 0; o >>= 1) v = fmaxf(v, __shfl_xor(v, o, 32));
  return v;
}

// ---------------------------------------------------------------------------
// Register-staged prefetch of one LDS tile (ROWS x BK halves).
//   load():  issue raw global loads into VGPRs (no waits consumed yet)
//   store(): convert f32->f16 if needed and ds_store_b128 into LDS
// Splitting the two lets the WMMA block sit between issue and consume.
// ---------------------------------------------------------------------------
template <typename T, int ROWS, int BK, int BLOCK>
struct Prefetch {
  static constexpr int ITERS = (ROWS * BK) / (BLOCK * 8);
  static constexpr int SPAD  = BK + 8;     // row padding: aligned, bank-spread
  uint4 raw[ITERS][sizeof(T) == 4 ? 2 : 1];

  __device__ __forceinline__ void load(const T* __restrict__ g, int ld, int k0,
                                       int tid) {
#pragma unroll
    for (int it = 0; it < ITERS; ++it) {
      const int e = (it * BLOCK + tid) * 8;
      const int r = e / BK;
      const int c = e - r * BK;
      const T* srcp = g + (size_t)r * ld + k0 + c;
      if constexpr (sizeof(T) == 4) {
        const uint4* p = reinterpret_cast<const uint4*>(srcp);
        raw[it][0] = p[0];
        raw[it][1] = p[1];
      } else {
        raw[it][0] = *reinterpret_cast<const uint4*>(srcp);
      }
    }
  }

  __device__ __forceinline__ void store(_Float16* __restrict__ s, int tid) {
#pragma unroll
    for (int it = 0; it < ITERS; ++it) {
      const int e = (it * BLOCK + tid) * 8;
      const int r = e / BK;
      const int c = e - r * BK;
      union { _Float16 h[8]; uint4 u; } t;
      if constexpr (sizeof(T) == 4) {
        union { uint4 u; float f[4]; } a, b;
        a.u = raw[it][0];
        b.u = raw[it][1];
        t.h[0] = (_Float16)a.f[0]; t.h[1] = (_Float16)a.f[1];
        t.h[2] = (_Float16)a.f[2]; t.h[3] = (_Float16)a.f[3];
        t.h[4] = (_Float16)b.f[0]; t.h[5] = (_Float16)b.f[1];
        t.h[6] = (_Float16)b.f[2]; t.h[7] = (_Float16)b.f[3];
      } else {
        t.u = raw[it][0];
      }
      *reinterpret_cast<uint4*>(&s[r * SPAD + c]) = t.u;
    }
  }
};

// ---------------------------------------------------------------------------
// WMMA GEMM: C[M,N] = A[M,K] @ W[N,K]^T (+bias)(gelu)(+residual), batched
// via blockIdx.z with element strides. 8 waves per 256-thread block laid
// out WM x WN; each wave owns ITM x ITN 16x16 accumulator tiles.
// Pipeline per K-step: issue global loads (k+1) -> WMMAs (k) -> cvt +
// ds_store (k+1) -> barrier.
// ---------------------------------------------------------------------------
template <int BM, int BN, int BK, int WM, int WN, typename TA, typename TB, int FLAGS>
__global__ __launch_bounds__(WM * WN * 32)
void gemm_wmma_kernel(const TA* __restrict__ A, int lda, long long strideA,
                      const TB* __restrict__ W, int ldb, long long strideB,
                      const float* __restrict__ bias,
                      const float* __restrict__ res,
                      float* __restrict__ C, int ldc, long long strideC,
                      int K) {
  constexpr int BLOCK = WM * WN * 32;
  constexpr int SPAD  = BK + 8;
  constexpr int WTM   = BM / WM;
  constexpr int WTN   = BN / WN;
  constexpr int ITM   = WTM / 16;
  constexpr int ITN   = WTN / 16;

  __shared__ _Float16 sA[2][BM * SPAD];
  __shared__ _Float16 sB[2][BN * SPAD];

  const int tid  = threadIdx.x;
  const int lane = tid & 31;
  const int wave = tid >> 5;
  const int wm   = wave % WM;
  const int wn   = wave / WM;
  const int l16  = lane & 15;
  const int half = lane >> 4;

  const int bm = blockIdx.x * BM;
  const int bn = blockIdx.y * BN;
  const long long batch = blockIdx.z;

  const TA* Ab = A + batch * strideA + (size_t)bm * lda;
  const TB* Wb = W + batch * strideB + (size_t)bn * ldb;
  float* Cb = C + batch * strideC;
  const float* Rb = (FLAGS & FLAG_RES) ? (res + batch * strideC) : nullptr;

  v8f acc[ITM][ITN];
  const v8f vzero = {0.f, 0.f, 0.f, 0.f, 0.f, 0.f, 0.f, 0.f};
#pragma unroll
  for (int i = 0; i < ITM; ++i)
#pragma unroll
    for (int j = 0; j < ITN; ++j) acc[i][j] = vzero;

  Prefetch<TA, BM, BK, BLOCK> pa;
  Prefetch<TB, BN, BK, BLOCK> pb;

  // Prologue: stage tile 0 into buffer 0.
  pa.load(Ab, lda, 0, tid);
  pb.load(Wb, ldb, 0, tid);
  pa.store(&sA[0][0], tid);
  pb.store(&sB[0][0], tid);
  __syncthreads();

  int buf = 0;
  for (int k0 = 0; k0 < K; k0 += BK, buf ^= 1) {
    const bool more = (k0 + BK) < K;
    // Issue next tile's global loads now; they complete under the WMMAs.
    if (more) {
      pa.load(Ab, lda, k0 + BK, tid);
      pb.load(Wb, ldb, k0 + BK, tid);
    }

    Frag fa[ITM], fb[ITN];
    // A fragment (16x32 f16): lanes 0-15 hold K 0-7 & 16-23, lanes 16-31
    // hold K 8-15 & 24-31 (ISA 7.12.2).
    const int ka = half * 8;
#pragma unroll
    for (int i = 0; i < ITM; ++i) {
      const int r = wm * WTM + i * 16 + l16;
      fa[i].u[0] = *reinterpret_cast<const uint4*>(&sA[buf][r * SPAD + ka]);
      fa[i].u[1] = *reinterpret_cast<const uint4*>(&sA[buf][r * SPAD + ka + 16]);
    }
    // B fragment (32x16 f16): same K order as A: lanes 0-15 hold K 0-7 &
    // 16-23, lanes 16-31 hold K 8-15 & 24-31.
    const int kb = half * 8;
#pragma unroll
    for (int j = 0; j < ITN; ++j) {
      const int n = wn * WTN + j * 16 + l16;
      fb[j].u[0] = *reinterpret_cast<const uint4*>(&sB[buf][n * SPAD + kb]);
      fb[j].u[1] = *reinterpret_cast<const uint4*>(&sB[buf][n * SPAD + kb + 16]);
    }
#pragma unroll
    for (int i = 0; i < ITM; ++i)
#pragma unroll
      for (int j = 0; j < ITN; ++j)
        acc[i][j] = __builtin_amdgcn_wmma_f32_16x16x32_f16(
            false, fa[i].v, false, fb[j].v, (short)0, acc[i][j], false, false);

    // Drain the prefetch into the alternate LDS buffer (s_wait_loadcnt lands
    // here, after the WMMA block has covered the load latency).
    if (more) {
      pa.store(&sA[buf ^ 1][0], tid);
      pb.store(&sB[buf ^ 1][0], tid);
    }
    // One barrier per K-step: this iteration's LDS reads are done before the
    // buffer flips, and the freshly staged tile is visible to all waves.
    __syncthreads();
  }

  // Epilogue. C/D layout: VGPR r -> M = r + 8*(lane>=16), N = lane&15.
#pragma unroll
  for (int i = 0; i < ITM; ++i) {
#pragma unroll
    for (int j = 0; j < ITN; ++j) {
      const int ng    = bn + wn * WTN + j * 16 + l16;
      const int mbase = bm + wm * WTM + i * 16 + half * 8;
      const float bv  = (FLAGS & FLAG_BIAS) ? bias[ng] : 0.f;
      float* cptr       = Cb + (size_t)mbase * ldc + ng;
      const float* rptr = (FLAGS & FLAG_RES) ? (Rb + (size_t)mbase * ldc + ng) : nullptr;
#pragma unroll
      for (int r = 0; r < 8; ++r) {
        float v = acc[i][j][r] + bv;
        if (FLAGS & FLAG_GELU) v = 0.5f * v * (1.f + erff(v * 0.70710678118f));
        if (FLAGS & FLAG_RES)  v += rptr[(size_t)r * ldc];
        cptr[(size_t)r * ldc] = v;
      }
    }
  }
}

// ---------------------------------------------------------------------------
// LayerNorm over D=1024 (one block per row, 256 threads)
// ---------------------------------------------------------------------------
__global__ __launch_bounds__(256)
void layernorm_kernel(const float* __restrict__ x, const float* __restrict__ g,
                      const float* __restrict__ b, float* __restrict__ y) {
  __shared__ float rs[8], rs2[8];
  const int row  = blockIdx.x;
  const int tid  = threadIdx.x;
  const int lane = tid & 31, wave = tid >> 5;
  const float* xr = x + (size_t)row * kD;
  float s = 0.f, s2 = 0.f;
#pragma unroll
  for (int j = tid; j < kD; j += 256) { const float v = xr[j]; s += v; s2 += v * v; }
  s = waveSum(s); s2 = waveSum(s2);
  if (lane == 0) { rs[wave] = s; rs2[wave] = s2; }
  __syncthreads();
  if (wave == 0) {
    float a  = (lane < 8) ? rs[lane]  : 0.f;
    float a2 = (lane < 8) ? rs2[lane] : 0.f;
    a = waveSum(a); a2 = waveSum(a2);
    if (lane == 0) { rs[0] = a; rs2[0] = a2; }
  }
  __syncthreads();
  const float mean = rs[0] / kD;
  const float var  = rs2[0] / kD - mean * mean;
  const float rstd = rsqrtf(var + kEps);
  float* yr = y + (size_t)row * kD;
#pragma unroll
  for (int j = tid; j < kD; j += 256) yr[j] = (xr[j] - mean) * rstd * g[j] + b[j];
}

// ---------------------------------------------------------------------------
// Row softmax over Sk=512 with optional fused causal mask; writes f16 P.
// ---------------------------------------------------------------------------
__global__ __launch_bounds__(256)
void softmax_kernel(const float* __restrict__ sc, _Float16* __restrict__ P,
                    int causal) {
  __shared__ float red[8];
  const int row  = blockIdx.x;        // bh*kS + q
  const int q    = row & (kS - 1);
  const int tid  = threadIdx.x;
  const int lane = tid & 31, wave = tid >> 5;
  const float* x = sc + (size_t)row * kS;
  _Float16*   p  = P  + (size_t)row * kS;
  const int limit = causal ? (q + 1) : kS;

  float lmax = -3.402823466e38f;
  for (int j = tid; j < kS; j += 256)
    if (j < limit) lmax = fmaxf(lmax, x[j]);
  lmax = waveMax(lmax);
  if (lane == 0) red[wave] = lmax;
  __syncthreads();
  if (wave == 0) {
    float v = (lane < 8) ? red[lane] : -3.402823466e38f;
    v = waveMax(v);
    if (lane == 0) red[0] = v;
  }
  __syncthreads();
  const float gmax = red[0];
  __syncthreads();

  float lsum = 0.f;
  for (int j = tid; j < kS; j += 256)
    if (j < limit) lsum += __expf(x[j] - gmax);
  lsum = waveSum(lsum);
  if (lane == 0) red[wave] = lsum;
  __syncthreads();
  if (wave == 0) {
    float v = (lane < 8) ? red[lane] : 0.f;
    v = waveSum(v);
    if (lane == 0) red[0] = v;
  }
  __syncthreads();
  const float inv = 1.f / red[0];
  for (int j = tid; j < kS; j += 256)
    p[j] = (j < limit) ? (_Float16)(__expf(x[j] - gmax) * inv) : (_Float16)0.f;
}

// ---------------------------------------------------------------------------
// Head gathers: [S,B,rowStride] f32 -> per-head f16 layouts for batched WMMA.
//   extract_head  : dst[(bh*S + s)*HD + d]   (Q rows / K rows, optional scale)
//   extract_headT : dst[(bh*HD + d)*S + s]   (V transposed for P@V as A.W^T)
// ---------------------------------------------------------------------------
__global__ __launch_bounds__(256)
void extract_head_kernel(const float* __restrict__ src, _Float16* __restrict__ dst,
                         int rowStride, int colOff, float scale) {
  const int idx = blockIdx.x * 256 + threadIdx.x;   // kBH*kS*kHD total
  const int d  = idx % kHD;
  const int s  = (idx / kHD) % kS;
  const int bh = idx / (kHD * kS);
  const int b  = bh / kH;
  const int h  = bh - b * kH;
  const float v = src[((size_t)s * kB + b) * rowStride + colOff + h * kHD + d];
  dst[idx] = (_Float16)(v * scale);
}

__global__ __launch_bounds__(256)
void extract_headT_kernel(const float* __restrict__ src, _Float16* __restrict__ dst,
                          int rowStride, int colOff) {
  const int idx = blockIdx.x * 256 + threadIdx.x;   // kBH*kHD*kS total
  const int s  = idx % kS;
  const int d  = (idx / kS) % kHD;
  const int bh = idx / (kS * kHD);
  const int b  = bh / kH;
  const int h  = bh - b * kH;
  dst[idx] = (_Float16)src[((size_t)s * kB + b) * rowStride + colOff + h * kHD + d];
}

// ---------------------------------------------------------------------------
// Host orchestration
// ---------------------------------------------------------------------------
extern "C" void kernel_launch(void* const* d_in, const int* in_sizes, int n_in,
                              void* d_out, int out_size, void* d_ws, size_t ws_size,
                              hipStream_t stream) {
  (void)in_sizes; (void)n_in; (void)out_size; (void)ws_size;

  const float* src       = (const float*)d_in[0];
  const float* tgt       = (const float*)d_in[1];
  const float* enc_qkv_w = (const float*)d_in[2];
  const float* enc_qkv_b = (const float*)d_in[3];
  const float* enc_out_w = (const float*)d_in[4];
  const float* enc_out_b = (const float*)d_in[5];
  const float* enc_ln1_g = (const float*)d_in[6];
  const float* enc_ln1_b = (const float*)d_in[7];
  const float* enc_fc1_w = (const float*)d_in[8];
  const float* enc_fc1_b = (const float*)d_in[9];
  const float* enc_fc2_w = (const float*)d_in[10];
  const float* enc_fc2_b = (const float*)d_in[11];
  const float* enc_ln2_g = (const float*)d_in[12];
  const float* enc_ln2_b = (const float*)d_in[13];
  const float* enc_norm_g = (const float*)d_in[14];
  const float* enc_norm_b = (const float*)d_in[15];
  const float* dec_qkv_w = (const float*)d_in[16];
  const float* dec_qkv_b = (const float*)d_in[17];
  const float* dec_sout_w = (const float*)d_in[18];
  const float* dec_sout_b = (const float*)d_in[19];
  const float* dec_ln1_g = (const float*)d_in[20];
  const float* dec_ln1_b = (const float*)d_in[21];
  const float* dec_q_w   = (const float*)d_in[22];
  const float* dec_q_b   = (const float*)d_in[23];
  const float* dec_kv_w  = (const float*)d_in[24];
  const float* dec_kv_b  = (const float*)d_in[25];
  const float* dec_cout_w = (const float*)d_in[26];
  const float* dec_cout_b = (const float*)d_in[27];
  const float* dec_ln2_g = (const float*)d_in[28];
  const float* dec_ln2_b = (const float*)d_in[29];
  const float* dec_fc1_w = (const float*)d_in[30];
  const float* dec_fc1_b = (const float*)d_in[31];
  const float* dec_fc2_w = (const float*)d_in[32];
  const float* dec_fc2_b = (const float*)d_in[33];
  const float* dec_ln3_g = (const float*)d_in[34];
  const float* dec_ln3_b = (const float*)d_in[35];
  const float* dec_norm_g = (const float*)d_in[36];
  const float* dec_norm_b = (const float*)d_in[37];
  float* out = (float*)d_out;

  // --- workspace carve-out (bump allocator, 256B aligned) ---
  char* ws = (char*)d_ws;
  size_t off = 0;
  auto alloc = [&](size_t bytes) -> char* {
    char* p = ws + off;
    off += (bytes + 255) & ~(size_t)255;
    return p;
  };
  float* lnbuf    = (float*)alloc((size_t)kM * kD * 4);        // LN outputs
  float* bigbuf   = (float*)alloc((size_t)kM * kFF * 4);       // qkv / kv / ffn hidden
  float* attn_out = (float*)alloc((size_t)kM * kD * 4);        // PV result in [s,b,h,d]
  float* xb0      = (float*)alloc((size_t)kM * kD * 4);
  float* xb1      = (float*)alloc((size_t)kM * kD * 4);
  float* yb0      = (float*)alloc((size_t)kM * kD * 4);
  float* yb1      = (float*)alloc((size_t)kM * kD * 4);
  float* yb2      = (float*)alloc((size_t)kM * kD * 4);
  float* memory   = (float*)alloc((size_t)kM * kD * 4);
  _Float16* qh    = (_Float16*)alloc((size_t)kBH * kS * kHD * 2);
  _Float16* kh    = (_Float16*)alloc((size_t)kBH * kS * kHD * 2);
  _Float16* vt    = (_Float16*)alloc((size_t)kBH * kHD * kS * 2);
  float* scores   = (float*)alloc((size_t)kBH * kS * kS * 4);
  _Float16* pbuf  = (_Float16*)alloc((size_t)kBH * kS * kS * 2);

  auto ln = [&](const float* x, const float* g, const float* b, float* y) {
    layernorm_kernel<<<kM, 256, 0, stream>>>(x, g, b, y);
  };
  auto gemm = [&](const float* A, int lda, const float* W, int ldb,
                  const float* bias, const float* res, float* C, int ldc,
                  int M, int N, int K, int flags) {
    dim3 grid(M / 128, N / 128, 1), blk(256);
    if (flags == FLAG_BIAS)
      gemm_wmma_kernel<128, 128, 32, 4, 2, float, float, FLAG_BIAS>
          <<<grid, blk, 0, stream>>>(A, lda, 0, W, ldb, 0, bias, nullptr, C, ldc, 0, K);
    else if (flags == (FLAG_BIAS | FLAG_GELU))
      gemm_wmma_kernel<128, 128, 32, 4, 2, float, float, FLAG_BIAS | FLAG_GELU>
          <<<grid, blk, 0, stream>>>(A, lda, 0, W, ldb, 0, bias, nullptr, C, ldc, 0, K);
    else
      gemm_wmma_kernel<128, 128, 32, 4, 2, float, float, FLAG_BIAS | FLAG_RES>
          <<<grid, blk, 0, stream>>>(A, lda, 0, W, ldb, 0, bias, res, C, ldc, 0, K);
  };
  constexpr int kExtractBlocks = (kBH * kS * kHD) / 256;
  auto attn_core = [&](int causal) {
    // scores[bh] = Q[bh] @ K[bh]^T   (M=N=512, K=64, 128 batches)
    gemm_wmma_kernel<128, 128, 32, 4, 2, _Float16, _Float16, 0>
        <<<dim3(kS / 128, kS / 128, kBH), 256, 0, stream>>>(
            qh, kHD, (long long)kS * kHD, kh, kHD, (long long)kS * kHD,
            nullptr, nullptr, scores, kS, (long long)kS * kS, kHD);
    softmax_kernel<<<kBH * kS, 256, 0, stream>>>(scores, pbuf, causal);
    // attn[bh] = P[bh] @ V[bh]; written straight into [s, b*H+h, d] layout
    gemm_wmma_kernel<128, 64, 32, 4, 2, _Float16, _Float16, 0>
        <<<dim3(kS / 128, 1, kBH), 256, 0, stream>>>(
            pbuf, kS, (long long)kS * kS, vt, kS, (long long)kHD * kS,
            nullptr, nullptr, attn_out, kB * kD, (long long)kHD, kS);
  };

  // ============================== encoder ==============================
  const float* xin = src;
  for (int i = 0; i < kL; ++i) {
    ln(xin, enc_ln1_g + (size_t)i * kD, enc_ln1_b + (size_t)i * kD, lnbuf);
    gemm(lnbuf, kD, enc_qkv_w + (size_t)i * 3 * kD * kD, kD,
         enc_qkv_b + (size_t)i * 3 * kD, nullptr, bigbuf, 3 * kD,
         kM, 3 * kD, kD, FLAG_BIAS);
    extract_head_kernel<<<kExtractBlocks, 256, 0, stream>>>(bigbuf, qh, 3 * kD, 0, kQScale);
    extract_head_kernel<<<kExtractBlocks, 256, 0, stream>>>(bigbuf, kh, 3 * kD, kD, 1.f);
    extract_headT_kernel<<<kExtractBlocks, 256, 0, stream>>>(bigbuf, vt, 3 * kD, 2 * kD);
    attn_core(/*causal=*/0);
    gemm(attn_out, kD, enc_out_w + (size_t)i * kD * kD, kD,
         enc_out_b + (size_t)i * kD, xin, xb0, kD, kM, kD, kD, FLAG_BIAS | FLAG_RES);
    ln(xb0, enc_ln2_g + (size_t)i * kD, enc_ln2_b + (size_t)i * kD, lnbuf);
    gemm(lnbuf, kD, enc_fc1_w + (size_t)i * kFF * kD, kD,
         enc_fc1_b + (size_t)i * kFF, nullptr, bigbuf, kFF,
         kM, kFF, kD, FLAG_BIAS | FLAG_GELU);
    gemm(bigbuf, kFF, enc_fc2_w + (size_t)i * kD * kFF, kFF,
         enc_fc2_b + (size_t)i * kD, xb0, xb1, kD, kM, kD, kFF, FLAG_BIAS | FLAG_RES);
    xin = xb1;
  }
  ln(xin, enc_norm_g, enc_norm_b, memory);

  // ============================== decoder ==============================
  const float* yin = tgt;
  for (int i = 0; i < kL; ++i) {
    // ---- masked self-attention ----
    ln(yin, dec_ln1_g + (size_t)i * kD, dec_ln1_b + (size_t)i * kD, lnbuf);
    gemm(lnbuf, kD, dec_qkv_w + (size_t)i * 3 * kD * kD, kD,
         dec_qkv_b + (size_t)i * 3 * kD, nullptr, bigbuf, 3 * kD,
         kM, 3 * kD, kD, FLAG_BIAS);
    extract_head_kernel<<<kExtractBlocks, 256, 0, stream>>>(bigbuf, qh, 3 * kD, 0, kQScale);
    extract_head_kernel<<<kExtractBlocks, 256, 0, stream>>>(bigbuf, kh, 3 * kD, kD, 1.f);
    extract_headT_kernel<<<kExtractBlocks, 256, 0, stream>>>(bigbuf, vt, 3 * kD, 2 * kD);
    attn_core(/*causal=*/1);
    gemm(attn_out, kD, dec_sout_w + (size_t)i * kD * kD, kD,
         dec_sout_b + (size_t)i * kD, yin, yb0, kD, kM, kD, kD, FLAG_BIAS | FLAG_RES);

    // ---- cross-attention ----
    ln(yb0, dec_ln2_g + (size_t)i * kD, dec_ln2_b + (size_t)i * kD, lnbuf);
    gemm(lnbuf, kD, dec_q_w + (size_t)i * kD * kD, kD,
         dec_q_b + (size_t)i * kD, nullptr, bigbuf, kD, kM, kD, kD, FLAG_BIAS);
    extract_head_kernel<<<kExtractBlocks, 256, 0, stream>>>(bigbuf, qh, kD, 0, kQScale);
    gemm(memory, kD, dec_kv_w + (size_t)i * 2 * kD * kD, kD,
         dec_kv_b + (size_t)i * 2 * kD, nullptr, bigbuf, 2 * kD,
         kM, 2 * kD, kD, FLAG_BIAS);
    extract_head_kernel<<<kExtractBlocks, 256, 0, stream>>>(bigbuf, kh, 2 * kD, 0, 1.f);
    extract_headT_kernel<<<kExtractBlocks, 256, 0, stream>>>(bigbuf, vt, 2 * kD, kD);
    attn_core(/*causal=*/0);
    gemm(attn_out, kD, dec_cout_w + (size_t)i * kD * kD, kD,
         dec_cout_b + (size_t)i * kD, yb0, yb1, kD, kM, kD, kD, FLAG_BIAS | FLAG_RES);

    // ---- FFN ----
    ln(yb1, dec_ln3_g + (size_t)i * kD, dec_ln3_b + (size_t)i * kD, lnbuf);
    gemm(lnbuf, kD, dec_fc1_w + (size_t)i * kFF * kD, kD,
         dec_fc1_b + (size_t)i * kFF, nullptr, bigbuf, kFF,
         kM, kFF, kD, FLAG_BIAS | FLAG_GELU);
    gemm(bigbuf, kFF, dec_fc2_w + (size_t)i * kD * kFF, kFF,
         dec_fc2_b + (size_t)i * kD, yb1, yb2, kD, kM, kD, kFF, FLAG_BIAS | FLAG_RES);
    yin = yb2;
  }
  ln(yin, dec_norm_g, dec_norm_b, out);
}
